// DCFNet_47897475285157
// MI455X (gfx1250) — compile-verified, matrix operand order checked by reading, not run on hardware
//
#include <hip/hip_runtime.h>
#include <hip/hip_bf16.h>

// ---------------------------------------------------------------------------
// DCFNet pipeline on MI455X (gfx1250, wave32, WMMA + Tensor Data Mover).
// Layout choices (all for wide loads/stores):
//   hidden  : NHWC f16  -> conv2 A-frag = two b128 loads + shuffle
//   weights : [t][co][ci] f16 in LDS -> B-frag = two ds b128 + shuffle
//   tables  : symmetric, staged row-major -> B-frag = one ds_load_b64
//   Y, G    : stored TRANSPOSED (k-major) by producer (contiguous v4f stores
//             from the C/D layout) -> consumer reads contiguous v2f pairs
// ---------------------------------------------------------------------------

#define S_   64
#define CIN_ 3
#define C_   32
#define H_   256
#define W_   256
#define WF_  129
#define WFP_ 144   // 9 tiles of 16; k>=129 carries zero contribution

typedef float    v2f  __attribute__((ext_vector_type(2)));
typedef float    v4f  __attribute__((ext_vector_type(4)));
typedef float    v8f  __attribute__((ext_vector_type(8)));
typedef _Float16 v8h  __attribute__((ext_vector_type(8)));
typedef _Float16 v16h __attribute__((ext_vector_type(16)));
typedef unsigned int u32x4 __attribute__((ext_vector_type(4)));
typedef int      i32x4 __attribute__((ext_vector_type(4)));
typedef int      i32x8 __attribute__((ext_vector_type(8)));

__device__ __forceinline__ v8f wmma4(v2f a, v2f b, v8f c) {
  return __builtin_amdgcn_wmma_f32_16x16x4_f32(false, a, false, b, (short)0, c,
                                               false, false);
}
__device__ __forceinline__ v8f wmma32(v16h a, v16h b, v8f c) {
  return __builtin_amdgcn_wmma_f32_16x16x32_f16(false, a, false, b, (short)0, c,
                                                false, false);
}
__device__ __forceinline__ v16h cat8(v8h a, v8h b) {
  return __builtin_shufflevector(a, b, 0, 1, 2, 3, 4, 5, 6, 7, 8, 9, 10, 11,
                                 12, 13, 14, 15);
}
__device__ __forceinline__ v2f ld2(const float* p) { return *(const v2f*)p; }
__device__ __forceinline__ unsigned lds_off(const void* p) {
  return (unsigned)(unsigned long long)p;   // LDS aperture: low 32b = offset
}

// TDM: load a [rows x tile0] tile (row stride stride_elems) into LDS.
// size_code: 0=1B 1=2B 2=4B. Descriptor packing per CDNA5 ISA §8.3/§8.4.
__device__ __forceinline__ void tdm_load_2d(const void* gsrc, unsigned ldsa,
                                            int size_code, int tile0, int rows,
                                            long long stride_elems) {
  unsigned long long ga = (unsigned long long)gsrc;
  u32x4 g0;
  g0.x = 1u;                                        // count=1, user D#
  g0.y = ldsa;                                      // lds_addr (bytes)
  g0.z = (unsigned)(ga & 0xffffffffull);            // global_addr lo
  g0.w = (unsigned)((ga >> 32) & 0x01ffffffull) | (2u << 30);  // hi | type=2
  i32x8 g1;
  g1[0] = size_code << 16;                          // data_size, mask=0
  g1[1] = (tile0 & 0xffff) << 16;                   // tensor_dim0 lo16
  g1[2] = ((tile0 >> 16) & 0xffff) | ((rows & 0xffff) << 16);  // d0 hi | d1 lo
  g1[3] = ((rows >> 16) & 0xffff) | ((tile0 & 0xffff) << 16);  // d1 hi | tile0
  g1[4] = rows & 0xffff;                            // tile_dim1 (tile_dim2=0)
  g1[5] = (int)(stride_elems & 0xffffffffll);       // dim0_stride lo32
  g1[6] = (int)((stride_elems >> 32) & 0xffffll);   // dim0_stride hi16
  g1[7] = 0;
  i32x4 gz4 = {0, 0, 0, 0};
  i32x8 gz8 = {0, 0, 0, 0, 0, 0, 0, 0};
  __builtin_amdgcn_tensor_load_to_lds(g0, g1, gz4, gz4, gz8, 0);
}

// ---------------------------------------------------------------------------
__global__ void __launch_bounds__(256) k_init(float* __restrict__ tc,
                                              float* __restrict__ ts,
                                              float* __restrict__ tn,
                                              const float* __restrict__ w2,
                                              _Float16* __restrict__ w2h) {
  int idx = blockIdx.x * 256 + threadIdx.x;   // 65536 threads
  int n = idx >> 8, k = idx & 255;
  float ph = 6.283185307179586f * (float)((n * k) & 255) * (1.0f / 256.0f);
  float c = __cosf(ph), s = __sinf(ph);
  tc[idx] = c; ts[idx] = s; tn[idx] = -s;
  if (idx < 9 * 32 * 32) {
    int t = idx >> 10, r = idx & 1023, ci = r >> 5, co = r & 31;
    // OIHW -> [tap][co][ci] so B-fragments are two contiguous 8-ci runs
    w2h[(t * 32 + co) * 32 + ci] = (_Float16)w2[(co * 32 + ci) * 9 + t];
  }
}

// ---------------------------------------------------------------------------
// conv1 + ReLU -> hidden in NHWC f16 (contiguous 32-ch pixel vectors).
__global__ void __launch_bounds__(256) k_conv1(const float* __restrict__ z,
                                               const float* __restrict__ w1,
                                               const float* __restrict__ b1,
                                               _Float16* __restrict__ hid) {
  int g = blockIdx.x * 256 + threadIdx.x;
  int s = g >> 16, rem = g & 65535, h = rem >> 8, w = rem & 255;
  float p[CIN_ * 9];
#pragma unroll
  for (int ci = 0; ci < CIN_; ++ci)
#pragma unroll
    for (int t = 0; t < 9; ++t) {
      int hh = h + t / 3 - 1, ww = w + t % 3 - 1;
      bool ok = ((unsigned)hh < (unsigned)H_) && ((unsigned)ww < (unsigned)W_);
      p[ci * 9 + t] =
          ok ? z[(((size_t)s * CIN_ + ci) * H_ + hh) * W_ + ww] : 0.0f;
    }
  _Float16* op = hid + ((size_t)g) * C_;   // NHWC: pixel-major
#pragma unroll 4
  for (int co = 0; co < C_; ++co) {
    float acc = b1[co];
#pragma unroll
    for (int q = 0; q < CIN_ * 9; ++q) acc += p[q] * w1[co * (CIN_ * 9) + q];
    op[co] = (_Float16)fmaxf(acc, 0.0f);
  }
}

// ---------------------------------------------------------------------------
// conv2 implicit GEMM (M=16 pixels, N=32 co, K=9 taps x 32 ci) + bias + LRN
// + cos_window. A-frag: two b128 NHWC loads; B-frag: two ds b128 from
// TDM-staged weights.
__global__ void __launch_bounds__(128) k_conv2(const _Float16* __restrict__ hid,
                                               const _Float16* __restrict__ w2h,
                                               const float* __restrict__ b2,
                                               const float* __restrict__ cw,
                                               float* __restrict__ xout) {
  __shared__ float sm[4][16][33];           // LRN staging
  __shared__ _Float16 smW[9 * 32 * 32];     // TDM-staged conv2 weights (18KB)
  int wave = threadIdx.x >> 5, lane = threadIdx.x & 31;
  int ln = lane & 15, lh = lane >> 4;
  int tile = blockIdx.x * 4 + wave;
  int w0 = (tile & 15) * 16;
  int h  = (tile >> 4) & 255;
  int s  = tile >> 12;

  if (wave == 0) {
    tdm_load_2d(w2h, lds_off(smW), /*2B*/ 1, 9 * 32 * 32, 1, 9 * 32 * 32);
    __builtin_amdgcn_s_wait_tensorcnt(0);
  }
  __syncthreads();

  v8f acc0 = {}, acc1 = {};
  v8h zero8 = {};
#pragma unroll
  for (int t = 0; t < 9; ++t) {
    int ky = t / 3 - 1, kx = t % 3 - 1;
    int hh = h + ky;
    if ((unsigned)hh >= (unsigned)H_) continue;
    int ww = w0 + ln + kx;
    bool inw = (unsigned)ww < (unsigned)W_;
    const _Float16* ap =
        hid + (((size_t)s * H_ + hh) * W_ + (size_t)ww) * C_ + 8 * lh;
    v8h r0 = inw ? *(const v8h*)(ap)      : zero8;   // ci 8lh .. 8lh+7
    v8h r1 = inw ? *(const v8h*)(ap + 16) : zero8;   // ci 16+8lh .. 23+8lh
    v16h a = cat8(r0, r1);
    const _Float16* wp = smW + (t * 32 + ln) * 32 + 8 * lh;
    v16h bA = cat8(*(const v8h*)(wp), *(const v8h*)(wp + 16));
    const _Float16* wq = smW + (t * 32 + 16 + ln) * 32 + 8 * lh;
    v16h bB = cat8(*(const v8h*)(wq), *(const v8h*)(wq + 16));
    acc0 = wmma32(a, bA, acc0);
    acc1 = wmma32(a, bB, acc1);
  }
  float bb0 = b2[ln], bb1 = b2[16 + ln];
#pragma unroll
  for (int v = 0; v < 8; ++v) { acc0[v] += bb0; acc1[v] += bb1; }

#pragma unroll
  for (int v = 0; v < 8; ++v) {
    int m = v + 8 * lh;
    sm[wave][m][ln]      = acc0[v];
    sm[wave][m][16 + ln] = acc1[v];
  }
  __syncthreads();

#pragma unroll
  for (int v = 0; v < 8; ++v) {
    int m = v + 8 * lh;
    float cwv = cw[h * W_ + w0 + m];
#pragma unroll
    for (int half = 0; half < 2; ++half) {
      int co = half * 16 + ln;
      float win = 0.0f;
#pragma unroll
      for (int j = -2; j <= 2; ++j) {
        int cj = co + j;
        if ((unsigned)cj < 32u) { float q = sm[wave][m][cj]; win += q * q; }
      }
      float av = half ? acc1[v] : acc0[v];
      float xv = av * __powf(1.0f + 2e-5f * win, -0.75f) * cwv;
      xout[(((size_t)s * C_ + co) * H_ + h) * W_ + w0 + m] = xv;
    }
  }
}

// ---------------------------------------------------------------------------
// Row rFFT. Tables symmetric: stage ROWS k0..k0+15 (row-major) -> per-lane
// contiguous K-pairs (ds_load_b64). Y stored transposed (k-major, v4f).
__global__ void __launch_bounds__(128) k_rfft_w(const float* __restrict__ x,
                                                const float* __restrict__ tc,
                                                const float* __restrict__ tn,
                                                float* __restrict__ yre,
                                                float* __restrict__ yim) {
  __shared__ float smC[16 * 256];   // rows k0..k0+15 of cos  (16KB)
  __shared__ float smN[16 * 256];   // rows of -sin           (16KB)
  int wave = threadIdx.x >> 5, lane = threadIdx.x & 31;
  int ln = lane & 15, lh = lane >> 4;
  int hb = blockIdx.x & 3;
  int kt = (blockIdx.x >> 2) % 9;
  int sc = blockIdx.x / 36;
  int h0 = (hb * 4 + wave) * 16;
  int k0 = kt * 16;

  if (wave == 0) {
    tdm_load_2d(tc + (size_t)k0 * 256, lds_off(smC), 2, 256, 16, 256);
    tdm_load_2d(tn + (size_t)k0 * 256, lds_off(smN), 2, 256, 16, 256);
    __builtin_amdgcn_s_wait_tensorcnt(0);
  }
  __syncthreads();

  const float* xr = x + ((size_t)sc * H_ + h0 + ln) * W_;
  __builtin_prefetch((const void*)xr, 0, 1);
  const float* bcr = smC + ln * 256;   // T[k0+ln][*] == T[*][k0+ln]
  const float* bnr = smN + ln * 256;
  v8f are = {}, aim = {};
  for (int kw = 0; kw < W_; kw += 4) {
    int kk = kw + 2 * lh;
    v2f a  = ld2(xr + kk);
    v2f bc = ld2(bcr + kk);
    v2f bn = ld2(bnr + kk);
    are = wmma4(a, bc, are);   // + x*cos
    aim = wmma4(a, bn, aim);   // - x*sin
  }
  // transposed store: Y_t[(sc*WFP+k)*H + h]; 8 consecutive h per lane
  float* orr = yre + ((size_t)sc * WFP_ + k0 + ln) * H_ + h0 + 8 * lh;
  float* oii = yim + ((size_t)sc * WFP_ + k0 + ln) * H_ + h0 + 8 * lh;
  *(v4f*)orr       = __builtin_shufflevector(are, are, 0, 1, 2, 3);
  *(v4f*)(orr + 4) = __builtin_shufflevector(are, are, 4, 5, 6, 7);
  *(v4f*)oii       = __builtin_shufflevector(aim, aim, 0, 1, 2, 3);
  *(v4f*)(oii + 4) = __builtin_shufflevector(aim, aim, 4, 5, 6, 7);
}

// ---------------------------------------------------------------------------
// Column DFT + wf + channel-sum. Y_t chunks [16 k-rows x 64 h] TDM-pipelined,
// double buffered; all LDS reads are contiguous v2f. G stored transposed.
__global__ void __launch_bounds__(128) k_fft_h(const float* __restrict__ yre,
                                               const float* __restrict__ yim,
                                               const float* __restrict__ wf,
                                               const float* __restrict__ tc,
                                               const float* __restrict__ ts,
                                               const float* __restrict__ tn,
                                               float* __restrict__ gre,
                                               float* __restrict__ gim) {
  __shared__ float smY[2][2][16 * 64];   // [parity][re/im][krow*64+h] 16KB
  int wave = threadIdx.x >> 5, lane = threadIdx.x & 31;
  int ln = lane & 15, lh = lane >> 4;
  int mb = blockIdx.x & 3;
  int kt = (blockIdx.x >> 2) % 9;
  int s  = blockIdx.x / 36;
  int m0 = (mb * 4 + wave) * 16;
  int k0 = kt * 16;
  const float* tcr = tc + (m0 + ln) * 256;
  const float* tsr = ts + (m0 + ln) * 256;
  const float* tnr = tn + (m0 + ln) * 256;

  const int NST = C_ * 4;                // 32 channels x 4 chunks of 64 h
  if (wave == 0) {
    const float* gr = yre + ((size_t)(s * C_ + 0) * WFP_ + k0) * H_ + 0;
    const float* gi = yim + ((size_t)(s * C_ + 0) * WFP_ + k0) * H_ + 0;
    tdm_load_2d(gr, lds_off(&smY[0][0][0]), 2, 64, 16, H_);
    tdm_load_2d(gi, lds_off(&smY[0][1][0]), 2, 64, 16, H_);
  }

  v8f gr8 = {}, gi8 = {}, zr = {}, zi = {};
  for (int st = 0; st < NST; ++st) {
    int par = st & 1;
    if (wave == 0) {
      if (st + 1 < NST) {
        int c1 = (st + 1) >> 2, ch1 = (st + 1) & 3;
        const float* gr =
            yre + ((size_t)(s * C_ + c1) * WFP_ + k0) * H_ + ch1 * 64;
        const float* gi =
            yim + ((size_t)(s * C_ + c1) * WFP_ + k0) * H_ + ch1 * 64;
        tdm_load_2d(gr, lds_off(&smY[par ^ 1][0][0]), 2, 64, 16, H_);
        tdm_load_2d(gi, lds_off(&smY[par ^ 1][1][0]), 2, 64, 16, H_);
        __builtin_amdgcn_s_wait_tensorcnt(2);   // retire current stage's pair
      } else {
        __builtin_amdgcn_s_wait_tensorcnt(0);
      }
    }
    __syncthreads();

    int ch = st & 3, hbase = ch * 64;
    const float* bR = &smY[par][0][ln * 64];   // k-row ln, h contiguous
    const float* bI = &smY[par][1][ln * 64];
    for (int hh = 0; hh < 64; hh += 4) {
      int kk = hh + 2 * lh, gk = hbase + kk;
      v2f ac = ld2(tcr + gk);
      v2f as = ld2(tsr + gk);
      v2f an = ld2(tnr + gk);
      v2f br = ld2(bR + kk);
      v2f bi = ld2(bI + kk);
      zr = wmma4(ac, br, zr);   // + cos*Yre
      zr = wmma4(as, bi, zr);   // + sin*Yim
      zi = wmma4(ac, bi, zi);   // + cos*Yim
      zi = wmma4(an, br, zi);   // - sin*Yre
    }
    if (ch == 3) {                              // finished channel c
      int c = st >> 2;
#pragma unroll
      for (int v = 0; v < 8; ++v) {
        int m = m0 + v + 8 * lh, k = k0 + ln;
        float wr = 0.0f, wi = 0.0f;
        if (k < WF_) {
          size_t wo = (((size_t)c * H_ + m) * WF_ + k) * 2;
          wr = wf[wo]; wi = wf[wo + 1];
        }
        gr8[v] += wr * zr[v];
        gi8[v] -= wi * zi[v];
      }
      zr = (v8f){}; zi = (v8f){};
    }
    __syncthreads();
  }
  // transposed store: G_t[(s*WFP+k)*H + m]
  float* orr = gre + ((size_t)s * WFP_ + k0 + ln) * H_ + m0 + 8 * lh;
  float* oii = gim + ((size_t)s * WFP_ + k0 + ln) * H_ + m0 + 8 * lh;
  *(v4f*)orr       = __builtin_shufflevector(gr8, gr8, 0, 1, 2, 3);
  *(v4f*)(orr + 4) = __builtin_shufflevector(gr8, gr8, 4, 5, 6, 7);
  *(v4f*)oii       = __builtin_shufflevector(gi8, gi8, 0, 1, 2, 3);
  *(v4f*)(oii + 4) = __builtin_shufflevector(gi8, gi8, 4, 5, 6, 7);
}

// ---------------------------------------------------------------------------
// Inverse column DFT: u[h][k] = sum_m G[m][k] e^{+2pi i m h/256}. G_t gives
// contiguous v2f B-pairs; alpha_k/N^2 folded into store.
__global__ void __launch_bounds__(128) k_ifft_h(const float* __restrict__ gre,
                                                const float* __restrict__ gim,
                                                const float* __restrict__ tc,
                                                const float* __restrict__ ts,
                                                const float* __restrict__ tn,
                                                float* __restrict__ u1,
                                                float* __restrict__ u2) {
  int wave = threadIdx.x >> 5, lane = threadIdx.x & 31;
  int ln = lane & 15, lh = lane >> 4;
  int tile = blockIdx.x * 4 + wave;            // S * (H/16) * 9
  int k0 = (tile % 9) * 16;
  int t2 = tile / 9;
  int h0 = (t2 & 15) * 16;
  int s  = t2 >> 4;
  const float* tcr = tc + (h0 + ln) * 256;
  const float* tsr = ts + (h0 + ln) * 256;
  const float* tnr = tn + (h0 + ln) * 256;
  const float* gbr = gre + ((size_t)s * WFP_ + k0 + ln) * H_;   // m contiguous
  const float* gbi = gim + ((size_t)s * WFP_ + k0 + ln) * H_;

  v8f ure = {}, uim = {};
  for (int mm = 0; mm < H_; mm += 4) {
    int kk = mm + 2 * lh;
    v2f ac = ld2(tcr + kk);
    v2f as = ld2(tsr + kk);
    v2f an = ld2(tnr + kk);
    v2f br = ld2(gbr + kk);
    v2f bi = ld2(gbi + kk);
    ure = wmma4(ac, br, ure);
    ure = wmma4(an, bi, ure);
    uim = wmma4(ac, bi, uim);
    uim = wmma4(as, br, uim);
  }
  int k = k0 + ln;
  float scl = (k == 0 || k == 128) ? (1.0f / 65536.0f)
            : (k < WF_)            ? (2.0f / 65536.0f)
                                   : 0.0f;
#pragma unroll
  for (int v = 0; v < 8; ++v) {
    size_t o = ((size_t)s * H_ + h0 + v + 8 * lh) * WFP_ + k;
    u1[o] =  ure[v] * scl;
    u2[o] = -uim[v] * scl;
  }
}

// ---------------------------------------------------------------------------
// Inverse row transform. Symmetric tables: stage rows w0..w0+15 x 144 cols.
__global__ void __launch_bounds__(128) k_ifft_w(const float* __restrict__ u1,
                                                const float* __restrict__ u2,
                                                const float* __restrict__ tc,
                                                const float* __restrict__ ts,
                                                float* __restrict__ out) {
  __shared__ float smC[16 * WFP_];   // rows w0..w0+15, cols 0..143 (9KB)
  __shared__ float smS[16 * WFP_];
  int wave = threadIdx.x >> 5, lane = threadIdx.x & 31;
  int ln = lane & 15, lh = lane >> 4;
  int hb = blockIdx.x & 3;
  int wt = (blockIdx.x >> 2) & 15;
  int s  = blockIdx.x >> 6;
  int h0 = (hb * 4 + wave) * 16;
  int w0 = wt * 16;

  if (wave == 0) {
    tdm_load_2d(tc + (size_t)w0 * 256, lds_off(smC), 2, WFP_, 16, 256);
    tdm_load_2d(ts + (size_t)w0 * 256, lds_off(smS), 2, WFP_, 16, 256);
    __builtin_amdgcn_s_wait_tensorcnt(0);
  }
  __syncthreads();

  const float* u1r = u1 + ((size_t)s * H_ + h0 + ln) * WFP_;
  const float* u2r = u2 + ((size_t)s * H_ + h0 + ln) * WFP_;
  __builtin_prefetch((const void*)u1r, 0, 1);
  const float* bcr = smC + ln * WFP_;   // T[w0+ln][*] == T[*][w0+ln]
  const float* bsr = smS + ln * WFP_;
  v8f acc = {};
  for (int kb = 0; kb < WFP_; kb += 4) {
    int kk = kb + 2 * lh;
    v2f a1 = ld2(u1r + kk);
    v2f a2 = ld2(u2r + kk);
    v2f bc = ld2(bcr + kk);
    v2f bs = ld2(bsr + kk);
    acc = wmma4(a1, bc, acc);
    acc = wmma4(a2, bs, acc);
  }
#pragma unroll
  for (int v = 0; v < 8; ++v)
    out[(size_t)s * H_ * W_ + (h0 + v + 8 * lh) * W_ + w0 + ln] = acc[v];
}

// ---------------------------------------------------------------------------
extern "C" void kernel_launch(void* const* d_in, const int* in_sizes, int n_in,
                              void* d_out, int out_size, void* d_ws,
                              size_t ws_size, hipStream_t stream) {
  (void)in_sizes; (void)n_in; (void)out_size; (void)ws_size;
  const float* z  = (const float*)d_in[0];
  const float* cw = (const float*)d_in[1];
  const float* wf = (const float*)d_in[2];
  const float* w1 = (const float*)d_in[3];
  const float* b1 = (const float*)d_in[4];
  const float* w2 = (const float*)d_in[5];
  const float* b2 = (const float*)d_in[6];
  float* out = (float*)d_out;

  char* p = (char*)d_ws;
  auto take = [&](size_t bytes) {
    char* r = p;
    p += (bytes + 255) & ~(size_t)255;
    return r;
  };
  float*    TC  = (float*)take((size_t)65536 * 4);
  float*    TS  = (float*)take((size_t)65536 * 4);
  float*    TN  = (float*)take((size_t)65536 * 4);
  _Float16* W2H = (_Float16*)take((size_t)9 * 32 * 32 * 2);
  _Float16* HID = (_Float16*)take((size_t)S_ * H_ * W_ * C_ * 2);   // NHWC
  float*    X   = (float*)take((size_t)S_ * C_ * H_ * W_ * 4);
  float*    YRE = (float*)take((size_t)S_ * C_ * WFP_ * H_ * 4);    // k-major
  float*    YIM = (float*)take((size_t)S_ * C_ * WFP_ * H_ * 4);
  float*    GRE = (float*)take((size_t)S_ * WFP_ * H_ * 4);         // k-major
  float*    GIM = (float*)take((size_t)S_ * WFP_ * H_ * 4);
  float*    U1  = (float*)take((size_t)S_ * H_ * WFP_ * 4);
  float*    U2  = (float*)take((size_t)S_ * H_ * WFP_ * 4);

  k_init<<<256, 256, 0, stream>>>(TC, TS, TN, w2, W2H);
  k_conv1<<<(S_ * H_ * W_) / 256, 256, 0, stream>>>(z, w1, b1, HID);
  k_conv2<<<(S_ * H_ * (W_ / 16)) / 4, 128, 0, stream>>>(HID, W2H, b2, cw, X);
  k_rfft_w<<<S_ * C_ * 9 * 4, 128, 0, stream>>>(X, TC, TN, YRE, YIM);
  k_fft_h<<<S_ * 9 * 4, 128, 0, stream>>>(YRE, YIM, wf, TC, TS, TN, GRE, GIM);
  k_ifft_h<<<(S_ * (H_ / 16) * 9) / 4, 128, 0, stream>>>(GRE, GIM, TC, TS, TN,
                                                         U1, U2);
  k_ifft_w<<<S_ * (H_ / 16), 128, 0, stream>>>(U1, U2, TC, TS, out);
}
